// Bilstm_crf_60507499266465
// MI455X (gfx1250) — matrix operand order repaired and checked by reading, not compiled, not run on hardware
//
#include <hip/hip_runtime.h>
#include <hip/hip_bf16.h>
#include <math.h>

#define S_LEN 256
#define B_SZ  64
#define H_SZ  512
#define E_SZ  300
#define EP    320        // E padded to multiple of 32 for K-loop
#define G_SZ  2048       // 4*H
#define T_SZ  10
#define KC_X  (EP / 32)          // 10 K-chunks from input projection
#define KC_H  (H_SZ / 32)        // 16 K-chunks from recurrence
#define KC_TOT (KC_X + KC_H)     // 26

typedef __bf16 bf16_t;
typedef __attribute__((ext_vector_type(16))) __bf16 v16bf;
typedef __attribute__((ext_vector_type(8)))  __bf16 v8bf;
typedef __attribute__((ext_vector_type(8)))  float  v8f;

// ---------------------------------------------------------------------------
// A-fragment loader: 16-bit A 16x32, contiguous k order (B loads use the same):
// lane<16 : elems 0..15 = K kb..kb+15
// lane>=16: elems 0..15 = K kb+16..kb+31
// caller passes ap already offset by (kb + hi*16)
// ---------------------------------------------------------------------------
__device__ __forceinline__ v16bf load_a_frag(const bf16_t* ap) {
    v8bf a0 = *(const v8bf*)(ap);
    v8bf a1 = *(const v8bf*)(ap + 8);
    return __builtin_shufflevector(a0, a1, 0,1,2,3,4,5,6,7,8,9,10,11,12,13,14,15);
}

// ---------------------------------------------------------------------------
// init: zero h state (2 dirs x 2 parity buffers) and c state
// ---------------------------------------------------------------------------
__global__ void init_state_kernel(bf16_t* h_st, float* c_st) {
    int idx = blockIdx.x * blockDim.x + threadIdx.x;
    if (idx < 4 * B_SZ * H_SZ) h_st[idx] = (bf16_t)0.0f;
    if (idx < 2 * B_SZ * H_SZ) c_st[idx] = 0.0f;
}

// ---------------------------------------------------------------------------
// embedding gather + bf16 cast + transpose to [S][B][EP] (E padded w/ zeros)
// ---------------------------------------------------------------------------
__global__ void embed_kernel(const int* __restrict__ tokens,
                             const float* __restrict__ emb,
                             bf16_t* __restrict__ x_bf) {
    int sb = blockIdx.x;            // s*B + b
    int s = sb >> 6;
    int b = sb & 63;
    int e = threadIdx.x;            // 0..EP-1
    int tok = tokens[b * S_LEN + s];
    float v = (e < E_SZ) ? emb[(size_t)tok * E_SZ + e] : 0.0f;
    x_bf[(size_t)sb * EP + e] = (bf16_t)v;
}

// ---------------------------------------------------------------------------
// weight cast to bf16 with gate-interleaved row permutation:
// dst row r  <-  src row (r&3)*H + (r>>2), K padded Ksrc -> Kdst with zeros
// ---------------------------------------------------------------------------
__global__ void cast_weight_kernel(const float* __restrict__ src,
                                   bf16_t* __restrict__ dst,
                                   int Ksrc, int Kdst) {
    int idx = blockIdx.x * blockDim.x + threadIdx.x;
    int total = G_SZ * Kdst;
    if (idx >= total) return;
    int r = idx / Kdst;
    int k = idx - r * Kdst;
    int orig = (r & 3) * H_SZ + (r >> 2);
    float v = (k < Ksrc) ? src[(size_t)orig * Ksrc + k] : 0.0f;
    dst[idx] = (bf16_t)v;
}

// ---------------------------------------------------------------------------
// One LSTM time step, both directions (blockIdx.z).
// gates = [x_s | h] @ [Wih | Whh]^T + b  (K = 320 + 512), gate-interleaved
// columns, bf16 WMMA 16x16x32 with f32 accumulation, software-pipelined
// (double-buffered A/B fragments), then fused cell update through LDS.
// Block = 128 threads (4 waves); each wave owns a 64x16 output strip
// (4 hidden units x 4 gates). Grid.x = 32 -> 128 strips = 2048 columns.
// h state double-buffered on step parity.
// ---------------------------------------------------------------------------
__global__ __launch_bounds__(128)
void lstm_step_kernel(int step,
                      const bf16_t* __restrict__ x_bf,
                      const bf16_t* __restrict__ wih_f_p,
                      const bf16_t* __restrict__ whh_f_p,
                      const float*  __restrict__ b_f,
                      const bf16_t* __restrict__ wih_b_p,
                      const bf16_t* __restrict__ whh_b_p,
                      const float*  __restrict__ b_b,
                      const int*    __restrict__ lengths,
                      bf16_t* __restrict__ h_st,   // [2dir][2parity][B][H]
                      float*  __restrict__ c_st,   // [2dir][B][H]
                      bf16_t* __restrict__ h_out)  // [S][B][2H]
{
    const int lane = threadIdx.x & 31;
    const int wave = threadIdx.x >> 5;
    const int dir  = blockIdx.z;
    const int col  = lane & 15;      // N within tile / weight-row select
    const int hi   = lane >> 4;      // K-half select
    const int s    = (dir == 0) ? step : (S_LEN - 1 - step);
    const int par  = step & 1;

    const bf16_t* wih  = dir ? wih_b_p : wih_f_p;
    const bf16_t* whh  = dir ? whh_b_p : whh_f_p;
    const float*  bias = dir ? b_b : b_f;

    const int nstrip = blockIdx.x * 4 + wave;     // 0..127
    const int r = nstrip * 16 + col;              // permuted weight row
    const float bval = bias[(r & 3) * H_SZ + (r >> 2)];

    v8f acc[4];
    #pragma unroll
    for (int m = 0; m < 4; ++m)
        #pragma unroll
        for (int v = 0; v < 8; ++v) acc[m][v] = bval;

    // unified fragment address generators over the fused K range
    const bf16_t* xa  = x_bf + (size_t)s * B_SZ * EP;
    const bf16_t* ha  = h_st + (size_t)(dir * 2 + par) * B_SZ * H_SZ;
    const bf16_t* wbp = wih + (size_t)r * EP   + hi * 16;
    const bf16_t* wbh = whh + (size_t)r * H_SZ + hi * 16;

    auto a_ptr = [&](int kc, int m) -> const bf16_t* {
        if (kc < KC_X)
            return xa + (size_t)(m * 16 + col) * EP + kc * 32 + hi * 16;
        return ha + (size_t)(m * 16 + col) * H_SZ + (kc - KC_X) * 32 + hi * 16;
    };
    auto b_ptr = [&](int kc) -> const bf16_t* {
        if (kc < KC_X) return wbp + kc * 32;
        return wbh + (kc - KC_X) * 32;
    };

    // ---- software-pipelined K loop: loads for kc+1 in flight during kc ----
    v16bf bcur = *(const v16bf*)b_ptr(0);
    v16bf acur[4];
    #pragma unroll
    for (int m = 0; m < 4; ++m) acur[m] = load_a_frag(a_ptr(0, m));

    #pragma unroll
    for (int kc = 0; kc < KC_TOT; ++kc) {
        v16bf bnxt = bcur;
        v16bf anxt[4] = { acur[0], acur[1], acur[2], acur[3] };
        if (kc + 1 < KC_TOT) {
            bnxt = *(const v16bf*)b_ptr(kc + 1);
            #pragma unroll
            for (int m = 0; m < 4; ++m) anxt[m] = load_a_frag(a_ptr(kc + 1, m));
        }
        if (kc + 2 < KC_TOT)
            __builtin_prefetch(b_ptr(kc + 2), 0, 3);   // global_prefetch weight stream
        #pragma unroll
        for (int m = 0; m < 4; ++m)
            acc[m] = __builtin_amdgcn_wmma_f32_16x16x32_bf16(
                false, acur[m], false, bcur, (short)0, acc[m], false, false);
        bcur = bnxt;
        #pragma unroll
        for (int m = 0; m < 4; ++m) acur[m] = anxt[m];
    }

    // ---- spill gates strip to LDS (C/D layout: M = v + hi*8, N = col) ----
    __shared__ float lds[64 * 68];
    #pragma unroll
    for (int m = 0; m < 4; ++m)
        #pragma unroll
        for (int v = 0; v < 8; ++v) {
            int brow = m * 16 + v + hi * 8;
            lds[brow * 68 + wave * 16 + col] = acc[m][v];
        }
    __syncthreads();

    // ---- fused LSTM cell update; this block owns 16 hidden units ----
    for (int p = threadIdx.x; p < B_SZ * 16; p += 128) {
        int b  = p >> 4;
        int jl = p & 15;
        float gi = lds[b * 68 + jl * 4 + 0];
        float gf = lds[b * 68 + jl * 4 + 1];
        float gg = lds[b * 68 + jl * 4 + 2];
        float go = lds[b * 68 + jl * 4 + 3];
        int jg = blockIdx.x * 16 + jl;

        size_t cidx = ((size_t)dir * B_SZ + b) * H_SZ + jg;
        float c_old = c_st[cidx];
        float ig = 1.0f / (1.0f + __expf(-gi));
        float fg = 1.0f / (1.0f + __expf(-gf));
        float og = 1.0f / (1.0f + __expf(-go));
        float cn = fg * c_old + ig * tanhf(gg);
        float hn = og * tanhf(cn);

        bool mk = s < lengths[b];
        size_t hr = (size_t)(dir * 2 + par)        * B_SZ * H_SZ + (size_t)b * H_SZ + jg;
        size_t hw = (size_t)(dir * 2 + (par ^ 1))  * B_SZ * H_SZ + (size_t)b * H_SZ + jg;
        bf16_t h_old = h_st[hr];
        c_st[cidx] = mk ? cn : c_old;
        h_st[hw]   = mk ? (bf16_t)hn : h_old;
        h_out[((size_t)s * B_SZ + b) * (2 * H_SZ) + dir * H_SZ + jg] =
            mk ? (bf16_t)hn : (bf16_t)0.0f;
    }
}

// ---------------------------------------------------------------------------
// feats[b][s][t] = h_cat[s][b][:] . w_out[t][:] + b_out[t]   (T=10, tiny)
// ---------------------------------------------------------------------------
__global__ void proj_kernel(const bf16_t* __restrict__ h_out,
                            const float* __restrict__ w_out,
                            const float* __restrict__ b_out,
                            float* __restrict__ em) {
    int idx = blockIdx.x * blockDim.x + threadIdx.x;
    if (idx >= B_SZ * S_LEN * T_SZ) return;
    int t  = idx % T_SZ;
    int sb = idx / T_SZ;
    int ss = sb % S_LEN;
    int b  = sb / S_LEN;
    const bf16_t* hp = h_out + ((size_t)ss * B_SZ + b) * (2 * H_SZ);
    const float*  wp = w_out + (size_t)t * (2 * H_SZ);
    float acc = b_out[t];
    for (int k = 0; k < 2 * H_SZ; k += 8) {
        v8bf hv = *(const v8bf*)(hp + k);
        #pragma unroll
        for (int u = 0; u < 8; ++u) acc += (float)hv[u] * wp[k + u];
    }
    em[((size_t)b * S_LEN + ss) * T_SZ + t] = acc;
}

// ---------------------------------------------------------------------------
// Viterbi decode: one thread per batch element (B=64, T=10)
// ---------------------------------------------------------------------------
__global__ void viterbi_kernel(const float* __restrict__ em,
                               const int*   __restrict__ lengths,
                               const float* __restrict__ startv,
                               const float* __restrict__ endv,
                               const float* __restrict__ trans,
                               int* __restrict__ bps,
                               int* __restrict__ out) {
    int b = threadIdx.x;
    if (b >= B_SZ) return;
    int len = lengths[b];

    float score[T_SZ];
    #pragma unroll
    for (int t = 0; t < T_SZ; ++t)
        score[t] = startv[t] + em[(size_t)b * S_LEN * T_SZ + t];

    int* bp = bps + (size_t)b * S_LEN * T_SZ;
    for (int ss = 1; ss < S_LEN; ++ss) {
        bool mk = ss < len;
        const float* emt = em + ((size_t)b * S_LEN + ss) * T_SZ;
        float ns[T_SZ]; int nb[T_SZ];
        for (int t = 0; t < T_SZ; ++t) {
            float best = score[0] + trans[t];
            int bi = 0;
            for (int p2 = 1; p2 < T_SZ; ++p2) {
                float cn = score[p2] + trans[p2 * T_SZ + t];
                if (cn > best) { best = cn; bi = p2; }   // first-max like argmax
            }
            ns[t] = mk ? (best + emt[t]) : score[t];
            nb[t] = mk ? bi : t;
        }
        for (int t = 0; t < T_SZ; ++t) { score[t] = ns[t]; bp[ss * T_SZ + t] = nb[t]; }
    }

    float bestf = score[0] + endv[0];
    int last = 0;
    for (int t = 1; t < T_SZ; ++t) {
        float cn = score[t] + endv[t];
        if (cn > bestf) { bestf = cn; last = t; }
    }

    int tag = last;
    out[b * S_LEN + (S_LEN - 1)] = ((S_LEN - 1) < len) ? tag : 0;
    for (int ss = S_LEN - 1; ss >= 1; --ss) {
        tag = bp[ss * T_SZ + tag];
        out[b * S_LEN + (ss - 1)] = ((ss - 1) < len) ? tag : 0;
    }
}

// ---------------------------------------------------------------------------
extern "C" void kernel_launch(void* const* d_in, const int* in_sizes, int n_in,
                              void* d_out, int out_size, void* d_ws, size_t ws_size,
                              hipStream_t stream) {
    (void)in_sizes; (void)n_in; (void)out_size; (void)ws_size;

    const int*   tokens  = (const int*)d_in[0];
    const int*   lengths = (const int*)d_in[1];
    const float* emb     = (const float*)d_in[2];
    const float* wih_f   = (const float*)d_in[3];
    const float* whh_f   = (const float*)d_in[4];
    const float* b_f     = (const float*)d_in[5];
    const float* wih_b   = (const float*)d_in[6];
    const float* whh_b   = (const float*)d_in[7];
    const float* b_b     = (const float*)d_in[8];
    const float* w_out   = (const float*)d_in[9];
    const float* b_out   = (const float*)d_in[10];
    const float* startv  = (const float*)d_in[11];
    const float* endv    = (const float*)d_in[12];
    const float* trans   = (const float*)d_in[13];

    // ---- workspace carve (all regions 256B aligned; total ~53 MB) ----
    char* ws = (char*)d_ws;
    size_t off = 0;
    auto carve = [&](size_t bytes) -> char* {
        char* p = ws + off;
        off += (bytes + 255) & ~(size_t)255;
        return p;
    };
    bf16_t* x_bf   = (bf16_t*)carve((size_t)S_LEN * B_SZ * EP * 2);       // 10.5 MB
    bf16_t* wihp_f = (bf16_t*)carve((size_t)G_SZ * EP * 2);               // 1.3 MB
    bf16_t* wihp_b = (bf16_t*)carve((size_t)G_SZ * EP * 2);
    bf16_t* whhp_f = (bf16_t*)carve((size_t)G_SZ * H_SZ * 2);             // 2.1 MB
    bf16_t* whhp_b = (bf16_t*)carve((size_t)G_SZ * H_SZ * 2);
    bf16_t* h_st   = (bf16_t*)carve((size_t)4 * B_SZ * H_SZ * 2);         // 2dir x 2par
    float*  c_st   = (float*) carve((size_t)2 * B_SZ * H_SZ * 4);
    bf16_t* h_out  = (bf16_t*)carve((size_t)S_LEN * B_SZ * 2 * H_SZ * 2); // 33.5 MB
    float*  em     = (float*) carve((size_t)B_SZ * S_LEN * T_SZ * 4);
    int*    bps    = (int*)   carve((size_t)B_SZ * S_LEN * T_SZ * 4);

    int* out = (int*)d_out;   // reference output dtype is int32 tags

    // ---- 1) zero recurrent state ----
    init_state_kernel<<<dim3((4 * B_SZ * H_SZ + 255) / 256), 256, 0, stream>>>(h_st, c_st);

    // ---- 2) embedding gather -> bf16 [S][B][EP] ----
    embed_kernel<<<dim3(S_LEN * B_SZ), dim3(EP), 0, stream>>>(tokens, emb, x_bf);

    // ---- 3) weight casts with gate-interleave permutation ----
    {
        int tot_ih = G_SZ * EP, tot_hh = G_SZ * H_SZ;
        cast_weight_kernel<<<dim3((tot_ih + 255) / 256), 256, 0, stream>>>(wih_f, wihp_f, E_SZ, EP);
        cast_weight_kernel<<<dim3((tot_ih + 255) / 256), 256, 0, stream>>>(wih_b, wihp_b, E_SZ, EP);
        cast_weight_kernel<<<dim3((tot_hh + 255) / 256), 256, 0, stream>>>(whh_f, whhp_f, H_SZ, H_SZ);
        cast_weight_kernel<<<dim3((tot_hh + 255) / 256), 256, 0, stream>>>(whh_b, whhp_b, H_SZ, H_SZ);
    }

    // ---- 4) 256 recurrent steps, fwd+bwd fused via grid.z ----
    for (int st = 0; st < S_LEN; ++st) {
        lstm_step_kernel<<<dim3(32, 1, 2), dim3(128), 0, stream>>>(
            st, x_bf, wihp_f, whhp_f, b_f, wihp_b, whhp_b, b_b,
            lengths, h_st, c_st, h_out);
    }

    // ---- 5) emission projection ----
    proj_kernel<<<dim3((B_SZ * S_LEN * T_SZ + 255) / 256), 256, 0, stream>>>(
        h_out, w_out, b_out, em);

    // ---- 6) Viterbi decode ----
    viterbi_kernel<<<dim3(1), dim3(64), 0, stream>>>(
        em, lengths, startv, endv, trans, bps, out);
}
